// MultiheadSelfAttention_2207613190171
// MI455X (gfx1250) — compile-verified, matrix operand order checked by reading, not run on hardware
//
#include <hip/hip_runtime.h>
#include <hip/hip_bf16.h>

// ---------------- types ----------------
typedef __attribute__((ext_vector_type(16))) __bf16 v16bf;
typedef __attribute__((ext_vector_type(8)))  float  v8f;

#define BB 4
#define SS 2048
#define DD 768
#define HH 12
#define HD 64
#define M_TOT (BB*SS)        // 8192
#define N_QKV (3*DD)         // 2304

__device__ __forceinline__ v8f wmma_bf16(v16bf a, v16bf b, v8f c) {
  // D = A(16x32) * B(32x16) + C, f32 accumulate
  return __builtin_amdgcn_wmma_f32_16x16x32_bf16(false, a, false, b, (short)0, c, false, false);
}

// Async global->LDS copy of 16 bytes per lane (CDNA5 VGLOBAL async op, ASYNCcnt).
// lds_byte = LDS byte offset (low 32 bits of generic pointer per flat-aperture rule).
__device__ __forceinline__ void async_copy16(unsigned lds_byte, const void* gptr) {
  asm volatile("global_load_async_to_lds_b128 %0, %1, off"
               :: "v"(lds_byte), "v"(gptr)
               : "memory");
}
#define WAIT_ASYNC_0() asm volatile("s_wait_asynccnt 0x0" ::: "memory")
#define WAIT_ASYNC_6() asm volatile("s_wait_asynccnt 0x6" ::: "memory")

__device__ __forceinline__ unsigned lds_off(const void* p) {
  return (unsigned)(unsigned long long)p;
}

// A/B fragment loader per CDNA5 ISA 7.12.2 (16-bit A 16x32 layout; B mirrored,
// source stored row-major [16 rows][K], K contiguous so each K-pair is one b32;
// adjacent pairs merge into ds_load_b128).
__device__ __forceinline__ v16bf load_frag(const __bf16* p, int ld, int row0, int k0) {
  const int lane = threadIdx.x & 31;
  const int m = lane & 15;
  const int g = (lane >> 4) & 1;           // half-wave selects K groups 0-7 / 8-15
  const __bf16* row = p + (row0 + m) * ld + k0 + g * 8;
  v16bf f;
#pragma unroll
  for (int v = 0; v < 8; ++v) {
    const int kb = ((v & 4) << 2) + 2 * (v & 3);   // v<4: 0,2,4,6 ; v>=4: 16,18,20,22
    f[2 * v]     = row[kb];
    f[2 * v + 1] = row[kb + 1];
  }
  return f;
}

__device__ __forceinline__ float xor16_max(float x) {
  x = fmaxf(x, __shfl_xor(x, 1, 32));
  x = fmaxf(x, __shfl_xor(x, 2, 32));
  x = fmaxf(x, __shfl_xor(x, 4, 32));
  x = fmaxf(x, __shfl_xor(x, 8, 32));
  return x;
}
__device__ __forceinline__ float xor16_sum(float x) {
  x += __shfl_xor(x, 1, 32);
  x += __shfl_xor(x, 2, 32);
  x += __shfl_xor(x, 4, 32);
  x += __shfl_xor(x, 8, 32);
  return x;
}

// ---------------- kernel 0: f32 -> bf16 ----------------
__global__ void cvt_f32_bf16(const float* __restrict__ in, __bf16* __restrict__ out, int n) {
  int i = blockIdx.x * blockDim.x + threadIdx.x;
  int stride = gridDim.x * blockDim.x;
  for (; i < n; i += stride) out[i] = (__bf16)in[i];
}

// ---------------- kernel 1: QKV projection ----------------
// X[8192,768] @ W[2304,768]^T -> scatter bf16 Q/K/V as [B,H,S,HD], Q scaled 1/8.
// 64x128 block tile, KT=64 per stage, double-buffered LDS fed by async loads:
// issue next tile (6 loads/thread) then s_wait_asynccnt<=6 -> previous tile done.
__global__ __launch_bounds__(256) void gemm_qkv(
    const __bf16* __restrict__ X, const __bf16* __restrict__ W,
    __bf16* __restrict__ Qb, __bf16* __restrict__ Kb, __bf16* __restrict__ Vb) {
  __shared__ __align__(16) __bf16 sA[2][64 * 64];
  __shared__ __align__(16) __bf16 sB[2][128 * 64];
  const int tid = threadIdx.x;
  const int lane = tid & 31;
  const int w = tid >> 5;
  const int wr = w >> 2, wc = w & 3;       // 2x4 wave grid, 32x32 per wave
  const int m0 = blockIdx.x * 64;
  const int n0 = blockIdx.y * 128;
  v8f acc[2][2] = {};

  auto issue = [&](int buf, int k0) {
#pragma unroll
    for (int i = 0; i < 2; ++i) {          // A tile: 64x64 = 512 chunks
      const int idx = tid + i * 256;
      const int r = idx >> 3, c = (idx & 7) * 8;
      async_copy16(lds_off(&sA[buf][r * 64 + c]), &X[(m0 + r) * DD + k0 + c]);
    }
#pragma unroll
    for (int i = 0; i < 4; ++i) {          // B tile: 128x64 = 1024 chunks
      const int idx = tid + i * 256;
      const int r = idx >> 3, c = (idx & 7) * 8;
      async_copy16(lds_off(&sB[buf][r * 64 + c]), &W[(n0 + r) * DD + k0 + c]);
    }
  };

  issue(0, 0);
  for (int kt = 0; kt < DD / 64; ++kt) {
    const int cur = kt & 1;
    if (kt + 1 < DD / 64) { issue(1 - cur, (kt + 1) * 64); WAIT_ASYNC_6(); }
    else                  { WAIT_ASYNC_0(); }
    __syncthreads();
    const __bf16* A  = sA[cur];
    const __bf16* Bt = sB[cur];
#pragma unroll
    for (int kk = 0; kk < 64; kk += 32) {
      v16bf a0 = load_frag(A, 64, wr * 32, kk);
      v16bf a1 = load_frag(A, 64, wr * 32 + 16, kk);
      v16bf b0 = load_frag(Bt, 64, wc * 32, kk);
      v16bf b1 = load_frag(Bt, 64, wc * 32 + 16, kk);
      acc[0][0] = wmma_bf16(a0, b0, acc[0][0]);
      acc[0][1] = wmma_bf16(a0, b1, acc[0][1]);
      acc[1][0] = wmma_bf16(a1, b0, acc[1][0]);
      acc[1][1] = wmma_bf16(a1, b1, acc[1][1]);
    }
    __syncthreads();                       // all waves done reading before re-issue
  }

  const int hi = (lane >> 4) & 1;
#pragma unroll
  for (int i = 0; i < 2; ++i)
#pragma unroll
    for (int j = 0; j < 2; ++j) {
      const int mt = m0 + wr * 32 + i * 16 + (hi << 3);
      const int nt = n0 + wc * 32 + j * 16 + (lane & 15);
      const int which = nt / DD;
      const int d = nt - which * DD;
      const int h = d >> 6, hd = d & 63;
#pragma unroll
      for (int r = 0; r < 8; ++r) {
        const int m = mt + r;
        const int b = m >> 11, s = m & 2047;
        const long dst = (((long)(b * HH + h)) * SS + s) * HD + hd;
        const float val = acc[i][j][r];
        if (which == 0)      Qb[dst] = (__bf16)(val * 0.125f);   // 1/sqrt(64)
        else if (which == 1) Kb[dst] = (__bf16)val;
        else                 Vb[dst] = (__bf16)val;
      }
    }
}

// ---------------- kernel 2: flash attention ----------------
// grid (S/64, H, B); 4 waves; wave w owns q rows [w*16, w*16+16)
__global__ __launch_bounds__(128) void flash_attn(
    const __bf16* __restrict__ Qb, const __bf16* __restrict__ Kb,
    const __bf16* __restrict__ Vb, __bf16* __restrict__ Ob) {
  __shared__ __align__(16) __bf16 sK[64 * 64];    // [kv][hd]
  __shared__ __align__(16) __bf16 sVT[64 * 64];   // [hd][kv]
  __shared__ __align__(16) __bf16 sP[4][16 * 64]; // per-wave P strip [m][kv]
  const int tid = threadIdx.x;
  const int lane = tid & 31;
  const int w = tid >> 5;
  const int hi = (lane >> 4) & 1;
  const int qt = blockIdx.x, h = blockIdx.y, b = blockIdx.z;
  const long base = ((long)(b * HH + h)) * SS * HD;
  const int q0 = qt * 64 + w * 16;

  // Q A-fragments (K-dim = 64 -> two 16x32 frags), loaded once from global
  v16bf aq0, aq1;
  {
    const __bf16* qp = Qb + base + (long)q0 * HD;
    const int m = lane & 15;
#pragma unroll
    for (int v = 0; v < 8; ++v) {
      const int kb = ((v & 4) << 2) + 2 * (v & 3) + hi * 8;
      aq0[2 * v]     = qp[m * HD + kb];
      aq0[2 * v + 1] = qp[m * HD + kb + 1];
      aq1[2 * v]     = qp[m * HD + 32 + kb];
      aq1[2 * v + 1] = qp[m * HD + 32 + kb + 1];
    }
  }

  v8f accO[4] = {};
  float mrow[8], lrow[8];
#pragma unroll
  for (int r = 0; r < 8; ++r) { mrow[r] = -1e30f; lrow[r] = 0.f; }

  for (int jt = 0; jt < SS / 64; ++jt) {
    __syncthreads();
    const __bf16* kg = Kb + base + (long)jt * 64 * HD;
    const __bf16* vg = Vb + base + (long)jt * 64 * HD;
#pragma unroll
    for (int i = 0; i < 4; ++i) {  // K tile: async DMA straight into LDS
      const int idx = tid + i * 128;
      const int r = idx >> 3, c = (idx & 7) * 8;
      async_copy16(lds_off(&sK[r * 64 + c]), &kg[r * 64 + c]);
    }
#pragma unroll
    for (int i = 0; i < 4; ++i) {  // V tile: vector load + transpose scatter
      const int idx = tid + i * 128;
      const int kv = idx >> 3, c = (idx & 7) * 8;
      uint4 dv = *(const uint4*)&vg[kv * 64 + c];
      const __bf16* e = (const __bf16*)&dv;
#pragma unroll
      for (int t = 0; t < 8; ++t) sVT[(c + t) * 64 + kv] = e[t];
    }
    WAIT_ASYNC_0();
    __syncthreads();

    // scores S = Q @ K^T : 4 column tiles of 16 kv each
    v8f sc[4];
#pragma unroll
    for (int n = 0; n < 4; ++n) {
      v8f z = {};
      z = wmma_bf16(aq0, load_frag(sK, 64, n * 16, 0), z);
      z = wmma_bf16(aq1, load_frag(sK, 64, n * 16, 32), z);
      sc[n] = z;
    }

    // online softmax (row stats live per half-wave, 8 rows per lane)
    float mi[8], alpha[8], rs[8];
#pragma unroll
    for (int r = 0; r < 8; ++r) {
      float mx = fmaxf(fmaxf(sc[0][r], sc[1][r]), fmaxf(sc[2][r], sc[3][r]));
      mx = xor16_max(mx);
      const float mn = fmaxf(mrow[r], mx);
      alpha[r] = __expf(mrow[r] - mn);
      mi[r] = mn;
      rs[r] = 0.f;
    }
#pragma unroll
    for (int n = 0; n < 4; ++n)
#pragma unroll
      for (int r = 0; r < 8; ++r) {
        const float p = __expf(sc[n][r] - mi[r]);
        sc[n][r] = p;
        rs[r] += p;
      }
#pragma unroll
    for (int r = 0; r < 8; ++r) {
      lrow[r] = lrow[r] * alpha[r] + xor16_sum(rs[r]);
      mrow[r] = mi[r];
    }
#pragma unroll
    for (int n = 0; n < 4; ++n)
#pragma unroll
      for (int r = 0; r < 8; ++r) accO[n][r] *= alpha[r];

    // D-layout -> A-fragment layout via per-wave LDS strip (DS ops in-order per wave)
    __bf16* pw = sP[w];
#pragma unroll
    for (int n = 0; n < 4; ++n)
#pragma unroll
      for (int r = 0; r < 8; ++r)
        pw[(r + hi * 8) * 64 + n * 16 + (lane & 15)] = (__bf16)sc[n][r];

    v16bf ap0 = load_frag(pw, 64, 0, 0);
    v16bf ap1 = load_frag(pw, 64, 0, 32);
#pragma unroll
    for (int n = 0; n < 4; ++n) {
      accO[n] = wmma_bf16(ap0, load_frag(sVT, 64, n * 16, 0), accO[n]);
      accO[n] = wmma_bf16(ap1, load_frag(sVT, 64, n * 16, 32), accO[n]);
    }
  }

  // epilogue: O / l -> bf16 [B,S,D] at head column h*64
  __bf16* op = Ob + ((long)b * SS + q0) * DD + h * HD;
#pragma unroll
  for (int n = 0; n < 4; ++n)
#pragma unroll
    for (int r = 0; r < 8; ++r) {
      const int row = r + hi * 8;
      op[(long)row * DD + n * 16 + (lane & 15)] = (__bf16)(accO[n][r] / lrow[r]);
    }
}

// ---------------- kernel 3: output projection ----------------
// O[8192,768] @ Wout[768,768]^T + b_out -> f32 out (same pipeline as gemm_qkv)
__global__ __launch_bounds__(256) void gemm_out(
    const __bf16* __restrict__ X, const __bf16* __restrict__ W,
    const float* __restrict__ bias, float* __restrict__ out) {
  __shared__ __align__(16) __bf16 sA[2][64 * 64];
  __shared__ __align__(16) __bf16 sB[2][128 * 64];
  const int tid = threadIdx.x;
  const int lane = tid & 31;
  const int w = tid >> 5;
  const int wr = w >> 2, wc = w & 3;
  const int m0 = blockIdx.x * 64;
  const int n0 = blockIdx.y * 128;
  v8f acc[2][2] = {};

  auto issue = [&](int buf, int k0) {
#pragma unroll
    for (int i = 0; i < 2; ++i) {
      const int idx = tid + i * 256;
      const int r = idx >> 3, c = (idx & 7) * 8;
      async_copy16(lds_off(&sA[buf][r * 64 + c]), &X[(m0 + r) * DD + k0 + c]);
    }
#pragma unroll
    for (int i = 0; i < 4; ++i) {
      const int idx = tid + i * 256;
      const int r = idx >> 3, c = (idx & 7) * 8;
      async_copy16(lds_off(&sB[buf][r * 64 + c]), &W[(n0 + r) * DD + k0 + c]);
    }
  };

  issue(0, 0);
  for (int kt = 0; kt < DD / 64; ++kt) {
    const int cur = kt & 1;
    if (kt + 1 < DD / 64) { issue(1 - cur, (kt + 1) * 64); WAIT_ASYNC_6(); }
    else                  { WAIT_ASYNC_0(); }
    __syncthreads();
    const __bf16* A  = sA[cur];
    const __bf16* Bt = sB[cur];
#pragma unroll
    for (int kk = 0; kk < 64; kk += 32) {
      v16bf a0 = load_frag(A, 64, wr * 32, kk);
      v16bf a1 = load_frag(A, 64, wr * 32 + 16, kk);
      v16bf b0 = load_frag(Bt, 64, wc * 32, kk);
      v16bf b1 = load_frag(Bt, 64, wc * 32 + 16, kk);
      acc[0][0] = wmma_bf16(a0, b0, acc[0][0]);
      acc[0][1] = wmma_bf16(a0, b1, acc[0][1]);
      acc[1][0] = wmma_bf16(a1, b0, acc[1][0]);
      acc[1][1] = wmma_bf16(a1, b1, acc[1][1]);
    }
    __syncthreads();
  }

  const int hi = (lane >> 4) & 1;
#pragma unroll
  for (int i = 0; i < 2; ++i)
#pragma unroll
    for (int j = 0; j < 2; ++j) {
      const int mt = m0 + wr * 32 + i * 16 + (hi << 3);
      const int nt = n0 + wc * 32 + j * 16 + (lane & 15);
      const float bv = bias[nt];
#pragma unroll
      for (int r = 0; r < 8; ++r)
        out[(long)(mt + r) * DD + nt] = acc[i][j][r] + bv;
    }
}

// ---------------- launcher ----------------
extern "C" void kernel_launch(void* const* d_in, const int* in_sizes, int n_in,
                              void* d_out, int out_size, void* d_ws, size_t ws_size,
                              hipStream_t stream) {
  const float* query = (const float*)d_in[0];   // [B,S,D]
  const float* w_qkv = (const float*)d_in[1];   // [3D,D]
  const float* w_out = (const float*)d_in[2];   // [D,D]
  const float* b_out = (const float*)d_in[3];   // [D]
  float* out = (float*)d_out;

  char* ws = (char*)d_ws;
  size_t off = 0;
  auto take = [&](size_t bytes) { void* p = ws + off; off += (bytes + 255) & ~(size_t)255; return p; };
  __bf16* Xbf  = (__bf16*)take((size_t)M_TOT * DD * 2);
  __bf16* Wqk  = (__bf16*)take((size_t)N_QKV * DD * 2);
  __bf16* Wo   = (__bf16*)take((size_t)DD * DD * 2);
  __bf16* Qb   = (__bf16*)take((size_t)M_TOT * DD * 2);
  __bf16* Kb   = (__bf16*)take((size_t)M_TOT * DD * 2);
  __bf16* Vb   = (__bf16*)take((size_t)M_TOT * DD * 2);
  __bf16* Ob   = (__bf16*)take((size_t)M_TOT * DD * 2);

  cvt_f32_bf16<<<2048, 256, 0, stream>>>(query, Xbf, M_TOT * DD);
  cvt_f32_bf16<<<1024, 256, 0, stream>>>(w_qkv, Wqk, N_QKV * DD);
  cvt_f32_bf16<<<512, 256, 0, stream>>>(w_out, Wo, DD * DD);

  gemm_qkv<<<dim3(M_TOT / 64, N_QKV / 128), 256, 0, stream>>>(Xbf, Wqk, Qb, Kb, Vb);
  flash_attn<<<dim3(SS / 64, HH, BB), 128, 0, stream>>>(Qb, Kb, Vb, Ob);
  gemm_out<<<dim3(M_TOT / 64, DD / 128), 256, 0, stream>>>(Ob, Wo, b_out, out);
}
